// SE3Score_28071906246775
// MI455X (gfx1250) — compile-verified, matrix operand order checked by reading, not run on hardware
//
#include <hip/hip_runtime.h>

// ---------------------------------------------------------------------------
// SE3Score for gfx1250 (MI455X).
//  - All dense GEMMs (q/k/v/self projections, conv weights, FC head) and the
//    per-edge radial MLP second layer (the FLOP-dominant block over 642k
//    cross edges) run on v_wmma_f32_16x16x32_f16.
//  - B operands are staged TRANSPOSED in LDS so both A and B fragment
//    gathers are two contiguous ds_load_b128 per lane (no u16 gathers).
//  - Segment softmax / scatter use f32 global atomics (monotonic-uint max).
// ---------------------------------------------------------------------------

typedef _Float16 half_t;
typedef __attribute__((ext_vector_type(16))) _Float16 v16h;
typedef __attribute__((ext_vector_type(8)))  float    v8f;

#define EPS_F 1e-8f

// Problem sizes (fixed by setup_inputs)
#define NR 1200
#define NL 250
#define ER 38400
#define EL 4000
#define NALL (NR + NL)
#define ECROSS ((long)ER + (long)EL + 2L * NR * NL) // 642400

union AFrag { v16h v; half_t h[16]; };
union CFrag { v8f  v; float  f[8];  };

// WMMA fragment K index for 16-bit A/B 16x16x32 tiles (per CDNA5 ISA layout).
__device__ __forceinline__ int frag_kk(int j, int lane) {
  return ((j >= 8) ? 16 : 0) + ((lane >= 16) ? 8 : 0) + (j & 7);
}

// ---------------------------------------------------------------------------
// Generic GEMM: C[M,N] = A[M,K] @ W[K,N] (+bias) (+relu). One wave per 16x16
// output tile; f32->f16 staging in LDS; B staged transposed; K % 32 == 0.
// ---------------------------------------------------------------------------
__global__ __launch_bounds__(32) void gemm_wmma_k(
    const float* __restrict__ A, const float* __restrict__ W,
    const float* __restrict__ bias, float* __restrict__ C,
    int M, int K, int N, int act)
{
  __shared__ half_t As[16 * 32];   // [m][kk]
  __shared__ half_t BsT[16 * 32];  // [n][kk]  (transposed stage)
  const int lane = threadIdx.x;
  const int tm = blockIdx.x * 16;
  const int tn = blockIdx.y * 16;

  CFrag acc;
#pragma unroll
  for (int i = 0; i < 8; ++i) acc.f[i] = 0.0f;

  for (int k0 = 0; k0 < K; k0 += 32) {
    if (lane < 16) {                       // A: one full 32-wide row per lane
      half_t* dstA = As + lane * 32;
      int r = tm + lane;
      if (r < M) {
        const float* src = A + (long)r * K + k0;
#pragma unroll
        for (int c = 0; c < 32; ++c) dstA[c] = (half_t)src[c];
      } else {
#pragma unroll
        for (int c = 0; c < 32; ++c) dstA[c] = (half_t)0.0f;
      }
    } else {                               // B: two k-rows per lane, scatter
      int kr = (lane - 16) * 2;
      for (int t = 0; t < 2; ++t) {
        const float* src = W + (long)(k0 + kr + t) * N + tn;
        if (tn + 16 <= N) {
#pragma unroll
          for (int nc = 0; nc < 16; ++nc)
            BsT[nc * 32 + kr + t] = (half_t)src[nc];
        } else {
          for (int nc = 0; nc < 16; ++nc)
            BsT[nc * 32 + kr + t] = (half_t)((tn + nc < N) ? src[nc] : 0.0f);
        }
      }
    }
    __syncthreads();

    AFrag af, bf;
    const int m = lane & 15;
#pragma unroll
    for (int j = 0; j < 16; ++j) {
      int kk = frag_kk(j, lane);
      af.h[j] = As[m * 32 + kk];    // two contiguous 8-half chunks
      bf.h[j] = BsT[m * 32 + kk];   // same pattern -> ds_load_b128 pairs
    }
    acc.v = __builtin_amdgcn_wmma_f32_16x16x32_f16(
        false, af.v, false, bf.v, (short)0, acc.v, false, false);
    __syncthreads();
  }

  const int n = lane & 15;
  const int mb = (lane >= 16) ? 8 : 0;
#pragma unroll
  for (int i = 0; i < 8; ++i) {
    int m = mb + i;
    if (tm + m < M && tn + n < N) {
      float v = acc.f[i];
      if (bias) v += bias[tn + n];
      if (act == 1) v = fmaxf(v, 0.0f);
      C[(long)(tm + m) * N + tn + n] = v;
    }
  }
}

// ---------------------------------------------------------------------------
// Helpers
// ---------------------------------------------------------------------------
__device__ __forceinline__ unsigned fkey(float f) {
  unsigned u = __float_as_uint(f);
  return (u & 0x80000000u) ? ~u : (u | 0x80000000u);
}
__device__ __forceinline__ float keyf(unsigned u) {
  u = (u & 0x80000000u) ? (u & 0x7fffffffu) : ~u;
  return __uint_as_float(u);
}

__device__ __forceinline__ void cross_edge(long e, const int* rs, const int* rd,
                                           const int* ls, const int* ld,
                                           int& s, int& d, int& t) {
  if (e < (long)ER)            { s = rs[e]; d = rd[e]; t = 0; }
  else if (e < (long)ER + EL)  { long i = e - ER; s = NR + ls[i]; d = NR + ld[i]; t = 1; }
  else {
    long c = e - ER - EL; t = 2;
    if (c < (long)NR * NL) { s = (int)(c / NL);       d = NR + (int)(c % NL); }
    else { c -= (long)NR * NL; s = NR + (int)(c % NL); d = (int)(c / NL); }
  }
}

// ---------------------------------------------------------------------------
// WMMA edge kernels. Block = 256 threads = 8 waves; each wave owns 16 edges.
// Per edge: h1 = relu(ef@rW1+rb1) (two lanes split the 32 hidden units),
// then rad[16 edges x C] = h1 @ rW2 via one WMMA per 16-channel tile.
// Dynamic LDS layout:
//   float sW1[rin*32], sb1[32], sb2[C], radt[8*256]
//   half  sW2t[C*32] (transposed: [c][j]), h1t[8*512]
// ---------------------------------------------------------------------------
#define EDGES_PER_BLOCK 128

struct EdgeCtx {
  int s, d;
  bool valid;
  float efv[6];
};

template <int CROSS>
__device__ __forceinline__ EdgeCtx edge_ctx(long e, long E, int rin,
    const float* __restrict__ ef, const float* __restrict__ xall,
    const int* esrc, const int* edst,
    const int* rs, const int* rd, const int* ls, const int* ld) {
  EdgeCtx cx;
  cx.valid = (e < E);
  cx.s = 0; cx.d = 0;
  for (int i = 0; i < 6; ++i) cx.efv[i] = 0.0f;
  if (!cx.valid) return cx;
  if (CROSS) {
    int t; cross_edge(e, rs, rd, ls, ld, cx.s, cx.d, t);
    float dx = xall[3 * cx.d + 0] - xall[3 * cx.s + 0];
    float dy = xall[3 * cx.d + 1] - xall[3 * cx.s + 1];
    float dz = xall[3 * cx.d + 2] - xall[3 * cx.s + 2];
    float r = sqrtf(dx * dx + dy * dy + dz * dz + EPS_F);
    cx.efv[0] = (t == 0) ? 1.0f : 0.0f;
    cx.efv[1] = (t == 1) ? 1.0f : 0.0f;
    cx.efv[2] = (t == 2) ? 1.0f : 0.0f;
    cx.efv[3] = r;
  } else {
    cx.s = esrc[e]; cx.d = edst[e];
    for (int i = 0; i < rin; ++i) cx.efv[i] = ef[e * rin + i];
  }
  return cx;
}

// MODE 0: attention logits (+ per-dst max). MODE 1: message/conv scatter.
template <int CROSS, int MODE>
__global__ __launch_bounds__(256) void edge_wmma_k(
    const float* __restrict__ ef, int rin, const float* __restrict__ xall,
    const int* __restrict__ esrc, const int* __restrict__ edst,
    const int* __restrict__ rs, const int* __restrict__ rd,
    const int* __restrict__ ls, const int* __restrict__ ld,
    long E, int C,
    const float* rW1, const float* rb1, const float* rW2, const float* rb2,
    const float* __restrict__ qv,   // MODE0: q  | MODE1: v
    const float* __restrict__ kx,   // MODE0: k  | MODE1: expv (nullable)
    const float* __restrict__ ssum, // MODE1 only (nullable)
    float* __restrict__ outA,       // MODE0: logits | MODE1: accumulate target
    unsigned* __restrict__ mkey)    // MODE0 only
{
  extern __shared__ float smem[];
  const int tid = threadIdx.x;
  const int w = tid >> 5, lane = tid & 31;
  const int m = lane & 15;          // edge slot within wave / frag row / col
  const int half = lane >> 4;       // which half of hidden units / channels

  float* sW1 = smem;
  float* sb1 = sW1 + rin * 32;
  float* sb2 = sb1 + 32;
  float* radt = sb2 + C;                       // 8*256 floats
  half_t* sW2t = (half_t*)(radt + 8 * 256);    // C*32 halves, [c][j]
  half_t* h1t = sW2t + 32 * C;                 // 8*512 halves

  {
    int nf = rin * 32 + 32 + C;
    for (int i = tid; i < nf; i += 256) {
      float v;
      if (i < rin * 32) v = rW1[i];
      else if (i < rin * 32 + 32) v = rb1[i - rin * 32];
      else v = rb2[i - rin * 32 - 32];
      smem[i] = v;
    }
    for (int i = tid; i < 32 * C; i += 256) {
      int j = i / C, c = i % C;                // contiguous read, scatter write
      sW2t[c * 32 + j] = (half_t)rW2[i];
    }
  }

  const long e = (long)blockIdx.x * EDGES_PER_BLOCK + w * 16 + m;
  EdgeCtx cx = edge_ctx<CROSS>(e, E, rin, ef, xall, esrc, edst, rs, rd, ls, ld);

  // h1 halves: lane pair (m, m+16) computes hidden units [half*16, half*16+16)
  {
    const int j0 = half * 16;
    half_t* dst = h1t + w * 512 + m * 32 + j0;
#pragma unroll
    for (int jj = 0; jj < 16; ++jj) {
      int j = j0 + jj;
      float a = sb1[j];
      for (int i = 0; i < rin; ++i) a += cx.efv[i] * sW1[i * 32 + j];
      dst[jj] = (half_t)fmaxf(a, 0.0f);
    }
  }
  __syncthreads();

  AFrag af;
#pragma unroll
  for (int j = 0; j < 16; ++j)
    af.h[j] = h1t[w * 512 + m * 32 + frag_kk(j, lane)];

  float a_w = 1.0f;
  if (MODE == 1 && kx && cx.valid) a_w = kx[e] / (ssum[cx.d] + EPS_F);

  const float* qd = qv + (long)cx.d * C;  // MODE0: q row
  const float* ks = (MODE == 0) ? (kx + (long)cx.s * C) : (qv + (long)cx.s * C);
  float* od = (MODE == 1) ? (outA + (long)cx.d * C) : nullptr;

  float lg = 0.0f;
  float* radw = radt + w * 256;
  const int ntiles = C >> 4;
  for (int ct = 0; ct < ntiles; ++ct) {
    AFrag bf;                                   // contiguous b128 pairs
#pragma unroll
    for (int j = 0; j < 16; ++j)
      bf.h[j] = sW2t[(ct * 16 + m) * 32 + frag_kk(j, lane)];

    CFrag acc;
#pragma unroll
    for (int i = 0; i < 8; ++i) acc.f[i] = 0.0f;
    acc.v = __builtin_amdgcn_wmma_f32_16x16x32_f16(
        false, af.v, false, bf.v, (short)0, acc.v, false, false);

    const int mb = (lane >= 16) ? 8 : 0;
#pragma unroll
    for (int i = 0; i < 8; ++i) radw[(mb + i) * 16 + m] = acc.f[i];
    __syncthreads();

    if (cx.valid) {
#pragma unroll
      for (int i = 0; i < 8; ++i) {
        int nn = half * 8 + i;
        int c = ct * 16 + nn;
        float rad = radw[m * 16 + nn] + sb2[c];
        if (MODE == 0) {
          lg += qd[c] * ks[c] * rad;
        } else {
          atomicAdd(&od[c], a_w * ks[c] * rad);
        }
      }
    }
    __syncthreads();
  }

  if (MODE == 0) {
    lg += __shfl_xor(lg, 16, 32);
    if (half == 0 && cx.valid) {
      lg *= rsqrtf((float)C);
      outA[e] = lg;
      atomicMax(&mkey[cx.d], fkey(lg));
    }
  }
}

// ---------------------------------------------------------------------------
// Pass B: e = exp(logit - max[dst]); sum per dst
// ---------------------------------------------------------------------------
template <int CROSS>
__global__ void edge_exp_k(
    const int* __restrict__ edst,
    const int* __restrict__ rs, const int* __restrict__ rd,
    const int* __restrict__ ls, const int* __restrict__ ld,
    long E, const float* __restrict__ logits,
    const unsigned* __restrict__ mkey,
    float* __restrict__ ssum, float* __restrict__ expv)
{
  long e = (long)blockIdx.x * blockDim.x + threadIdx.x;
  if (e >= E) return;
  int d;
  if (CROSS) { int s, t; cross_edge(e, rs, rd, ls, ld, s, d, t); }
  else d = edst[e];
  float m = keyf(mkey[d]);
  float ev = expf(logits[e] - m);
  expv[e] = ev;
  atomicAdd(&ssum[d], ev);
}

// ---------------------------------------------------------------------------
// Small utility kernels
// ---------------------------------------------------------------------------
__global__ void fill_f32_k(float* p, float v, long n) {
  long i = (long)blockIdx.x * blockDim.x + threadIdx.x;
  if (i < n) p[i] = v;
}
__global__ void fill_u32_k(unsigned* p, unsigned v, long n) {
  long i = (long)blockIdx.x * blockDim.x + threadIdx.x;
  if (i < n) p[i] = v;
}
__global__ void attn_fin_k(const float* __restrict__ h2,
                           const float* __restrict__ scale,
                           const float* __restrict__ bias,
                           float* __restrict__ out, long total, int C) {
  long i = (long)blockIdx.x * blockDim.x + threadIdx.x;
  if (i >= total) return;
  float x = h2[i];
  int c = (int)(i % C);
  float y = fmaxf(fabsf(x) * scale[c] + bias[c], 0.0f);
  float sg = (x > 0.0f) ? 1.0f : ((x < 0.0f) ? -1.0f : 0.0f);
  out[i] = y * sg;
}
__global__ void build_xall_k(const float* rx, const float* lx, float* xall) {
  int i = blockIdx.x * blockDim.x + threadIdx.x;
  if (i >= NALL * 3) return;
  int row = i / 3, c = i % 3;
  xall[i] = (row < NR) ? rx[i] : lx[(row - NR) * 3 + c];
}
__global__ void ef_rec_k(const int* src, const int* dst, const float* x,
                         const float* sv, const float* w, float* ef, long E) {
  long e = (long)blockIdx.x * blockDim.x + threadIdx.x;
  if (e >= E) return;
  int s = src[e], d = dst[e];
  float dx = x[3 * d] - x[3 * s], dy = x[3 * d + 1] - x[3 * s + 1],
        dz = x[3 * d + 2] - x[3 * s + 2];
  float r = sqrtf(dx * dx + dy * dy + dz * dz + EPS_F);
  float inv = 1.0f / r;
  ef[e * 6 + 0] = w[e * 4 + 0];
  ef[e * 6 + 1] = w[e * 4 + 1];
  ef[e * 6 + 2] = w[e * 4 + 2];
  ef[e * 6 + 3] = w[e * 4 + 3];
  ef[e * 6 + 4] = r;
  ef[e * 6 + 5] = (dx * sv[3 * s] + dy * sv[3 * s + 1] + dz * sv[3 * s + 2]) * inv;
}
__global__ void ef_lig_k(const int* src, const int* dst, const float* x,
                         const float* w, float* ef, long E) {
  long e = (long)blockIdx.x * blockDim.x + threadIdx.x;
  if (e >= E) return;
  int s = src[e], d = dst[e];
  float dx = x[3 * d] - x[3 * s], dy = x[3 * d + 1] - x[3 * s + 1],
        dz = x[3 * d + 2] - x[3 * s + 2];
  float r = sqrtf(dx * dx + dy * dy + dz * dz + EPS_F);
  ef[e * 5 + 0] = w[e * 4 + 0];
  ef[e * 5 + 1] = w[e * 4 + 1];
  ef[e * 5 + 2] = w[e * 4 + 2];
  ef[e * 5 + 3] = w[e * 4 + 3];
  ef[e * 5 + 4] = r;
}
__global__ void pool_mean_k(const float* h, float* g, int n, int C) {
  int c = threadIdx.x;
  if (c >= C) return;
  float s = 0.0f;
  for (int r = 0; r < n; ++r) s += h[(long)r * C + c];
  g[c] = s / (float)n;
}

// ---------------------------------------------------------------------------
// Host orchestration
// ---------------------------------------------------------------------------
struct LayerP { const float *Wk, *Wq, *Wself, *Wv, *bias, *rW1, *rb1, *rW2, *rb2, *scale; };
struct ConvP  { const float *Wself, *Wv, *bself, *rW1, *rb1, *rW2, *rb2; };

extern "C" void kernel_launch(void* const* d_in, const int* in_sizes, int n_in,
                              void* d_out, int out_size, void* d_ws, size_t ws_size,
                              hipStream_t stream) {
  (void)in_sizes; (void)n_in; (void)out_size; (void)ws_size;

  const int*   rec_src = (const int*)d_in[0];
  const int*   rec_dst = (const int*)d_in[1];
  const float* rec_x   = (const float*)d_in[2];
  const float* rec_f   = (const float*)d_in[3];
  const float* rec_sv  = (const float*)d_in[4];
  const float* rec_w   = (const float*)d_in[5];
  const int*   lig_src = (const int*)d_in[6];
  const int*   lig_dst = (const int*)d_in[7];
  const float* lig_x   = (const float*)d_in[8];
  const float* lig_f   = (const float*)d_in[9];
  const float* lig_w   = (const float*)d_in[10];

  // params: jax pytree order (sorted dict keys): cross, fc, lig, rec;
  // within a net: conv, layers; capital-letter keys sort first.
  int pi = 11;
  auto P = [&]() { return (const float*)d_in[pi++]; };
  ConvP  cC{P(), P(), P(), P(), P(), P(), P()};
  LayerP cL0{P(), P(), P(), P(), P(), P(), P(), P(), P(), P()};
  LayerP cL1{P(), P(), P(), P(), P(), P(), P(), P(), P(), P()};
  const float *W1 = P(), *b1 = P(), *W2 = P(), *b2 = P(), *W3 = P(), *b3 = P();
  ConvP  lC{P(), P(), P(), P(), P(), P(), P()};
  LayerP lL0{P(), P(), P(), P(), P(), P(), P(), P(), P(), P()};
  LayerP lL1{P(), P(), P(), P(), P(), P(), P(), P(), P(), P()};
  ConvP  rC{P(), P(), P(), P(), P(), P(), P()};
  LayerP rL0{P(), P(), P(), P(), P(), P(), P(), P(), P(), P()};
  LayerP rL1{P(), P(), P(), P(), P(), P(), P(), P(), P(), P()};

  float* F = (float*)d_ws;
  size_t off = 0;
  auto alloc = [&](size_t n) { float* p = F + off; off += n; return p; };
  float* xall   = alloc((size_t)NALL * 3);
  float* ef_rec = alloc((size_t)ER * 6);
  float* ef_lig = alloc((size_t)EL * 5);
  float* hA  = alloc((size_t)NALL * 128);
  float* hB  = alloc((size_t)NALL * 128);
  float* hC  = alloc((size_t)NALL * 128);
  float* qb  = alloc((size_t)NALL * 128);
  float* kb  = alloc((size_t)NALL * 128);
  float* vb  = alloc((size_t)NALL * 128);
  float* h2b = alloc((size_t)NALL * 128);
  float*    ssum = alloc(NALL);
  unsigned* mkey = (unsigned*)alloc(NALL);
  float* logits = alloc((size_t)ECROSS);
  float* expv   = alloc((size_t)ECROSS);
  float* g  = alloc(128);
  float* f1 = alloc(128);
  float* f2 = alloc(128);

  auto gemm = [&](const float* A, const float* W, const float* bias, float* C,
                  int M, int K, int N, int act) {
    dim3 gr((M + 15) / 16, (N + 15) / 16);
    gemm_wmma_k<<<gr, 32, 0, stream>>>(A, W, bias, C, M, K, N, act);
  };
  auto edge_smem = [&](int rin, int C) {
    return (size_t)(rin * 32 + 32 + C + 8 * 256) * 4 +
           (size_t)(32 * C + 8 * 512) * 2;
  };

  auto attn_layer = [&](const float* hin, float* hout, int Nn, int din, int Cc,
                        long E, int rin, const float* ef, bool cross,
                        const int* esrc, const int* edst, const LayerP& L) {
    gemm(hin, L.Wq, nullptr, qb, Nn, din, Cc, 0);
    gemm(hin, L.Wk, nullptr, kb, Nn, din, Cc, 0);
    gemm(hin, L.Wv, nullptr, vb, Nn, din, Cc, 0);
    gemm(hin, L.Wself, nullptr, h2b, Nn, din, Cc, 0);
    fill_u32_k<<<(Nn + 255) / 256, 256, 0, stream>>>(mkey, 0u, Nn);
    fill_f32_k<<<(Nn + 255) / 256, 256, 0, stream>>>(ssum, 0.0f, Nn);
    size_t sm = edge_smem(rin, Cc);
    unsigned gb = (unsigned)((E + EDGES_PER_BLOCK - 1) / EDGES_PER_BLOCK);
    unsigned gs = (unsigned)((E + 255) / 256);
    if (cross) {
      edge_wmma_k<1, 0><<<gb, 256, sm, stream>>>(nullptr, rin, xall, nullptr, nullptr,
          rec_src, rec_dst, lig_src, lig_dst, E, Cc,
          L.rW1, L.rb1, L.rW2, L.rb2, qb, kb, nullptr, logits, mkey);
      edge_exp_k<1><<<gs, 256, 0, stream>>>(nullptr, rec_src, rec_dst, lig_src, lig_dst,
          E, logits, mkey, ssum, expv);
      edge_wmma_k<1, 1><<<gb, 256, sm, stream>>>(nullptr, rin, xall, nullptr, nullptr,
          rec_src, rec_dst, lig_src, lig_dst, E, Cc,
          L.rW1, L.rb1, L.rW2, L.rb2, vb, expv, ssum, h2b, nullptr);
    } else {
      edge_wmma_k<0, 0><<<gb, 256, sm, stream>>>(ef, rin, nullptr, esrc, edst,
          nullptr, nullptr, nullptr, nullptr, E, Cc,
          L.rW1, L.rb1, L.rW2, L.rb2, qb, kb, nullptr, logits, mkey);
      edge_exp_k<0><<<gs, 256, 0, stream>>>(edst, nullptr, nullptr, nullptr, nullptr,
          E, logits, mkey, ssum, expv);
      edge_wmma_k<0, 1><<<gb, 256, sm, stream>>>(ef, rin, nullptr, esrc, edst,
          nullptr, nullptr, nullptr, nullptr, E, Cc,
          L.rW1, L.rb1, L.rW2, L.rb2, vb, expv, ssum, h2b, nullptr);
    }
    attn_fin_k<<<(unsigned)(((long)Nn * Cc + 255) / 256), 256, 0, stream>>>(
        h2b, L.scale, L.bias, hout, (long)Nn * Cc, Cc);
  };

  auto conv_layer = [&](const float* hin, float* hout, int Nn, int din, int dout,
                        long E, int rin, const float* ef, bool cross,
                        const int* esrc, const int* edst, const ConvP& Cv) {
    gemm(hin, Cv.Wv, nullptr, vb, Nn, din, dout, 0);
    gemm(hin, Cv.Wself, Cv.bself, hout, Nn, din, dout, 0); // edges add on top
    size_t sm = edge_smem(rin, dout);
    unsigned gb = (unsigned)((E + EDGES_PER_BLOCK - 1) / EDGES_PER_BLOCK);
    if (cross) {
      edge_wmma_k<1, 1><<<gb, 256, sm, stream>>>(nullptr, rin, xall, nullptr, nullptr,
          rec_src, rec_dst, lig_src, lig_dst, E, dout,
          Cv.rW1, Cv.rb1, Cv.rW2, Cv.rb2, vb, nullptr, nullptr, hout, nullptr);
    } else {
      edge_wmma_k<0, 1><<<gb, 256, sm, stream>>>(ef, rin, nullptr, esrc, edst,
          nullptr, nullptr, nullptr, nullptr, E, dout,
          Cv.rW1, Cv.rb1, Cv.rW2, Cv.rb2, vb, nullptr, nullptr, hout, nullptr);
    }
  };

  // ---- precompute ----
  build_xall_k<<<(NALL * 3 + 255) / 256, 256, 0, stream>>>(rec_x, lig_x, xall);
  ef_rec_k<<<(ER + 255) / 256, 256, 0, stream>>>(rec_src, rec_dst, rec_x, rec_sv, rec_w, ef_rec, ER);
  ef_lig_k<<<(EL + 255) / 256, 256, 0, stream>>>(lig_src, lig_dst, lig_x, lig_w, ef_lig, EL);

  // ---- rec net: 2 attn (32->32) + conv (32->64) ----
  attn_layer(rec_f, hA, NR, 32, 32, ER, 6, ef_rec, false, rec_src, rec_dst, rL0);
  attn_layer(hA,    hB, NR, 32, 32, ER, 6, ef_rec, false, rec_src, rec_dst, rL1);
  conv_layer(hB, hA, NR, 32, 64, ER, 6, ef_rec, false, rec_src, rec_dst, rC);

  // ---- lig net: 2 attn (32->32) + conv (32->64), output rows NR.. of hA ----
  attn_layer(lig_f, hB, NL, 32, 32, EL, 5, ef_lig, false, lig_src, lig_dst, lL0);
  attn_layer(hB,    hC, NL, 32, 32, EL, 5, ef_lig, false, lig_src, lig_dst, lL1);
  conv_layer(hC, hA + (size_t)NR * 64, NL, 32, 64, EL, 5, ef_lig, false, lig_src, lig_dst, lC);

  // ---- cross net: 2 attn (64->64) + conv (64->128) on combined graph ----
  attn_layer(hA, hB, NALL, 64, 64, ECROSS, 4, nullptr, true, nullptr, nullptr, cL0);
  attn_layer(hB, hC, NALL, 64, 64, ECROSS, 4, nullptr, true, nullptr, nullptr, cL1);
  conv_layer(hC, hB, NALL, 64, 128, ECROSS, 4, nullptr, true, nullptr, nullptr, cC);

  // ---- pooling + FC head ----
  pool_mean_k<<<1, 128, 0, stream>>>(hB, g, NALL, 128);
  gemm(g,  W1, b1, f1, 1, 128, 128, 1);
  gemm(f1, W2, b2, f2, 1, 128, 128, 1);
  gemm(f2, W3, b3, (float*)d_out, 1, 128, 2, 0);
}
